// AttentionGNN_48000554500179
// MI455X (gfx1250) — matrix operand order repaired and checked by reading, not compiled, not run on hardware
//
#include <hip/hip_runtime.h>
#include <math.h>

#define NNODE 100000
#define NEDGE 1200000

typedef __attribute__((ext_vector_type(16))) _Float16 v16h;
typedef __attribute__((ext_vector_type(8)))  float    v8f;

// ---------------------------------------------------------------------------
// f32x8 -> f16 conversion into fragment halves
// ---------------------------------------------------------------------------
__device__ __forceinline__ void cvt8(v16h& a, int base, float4 f0, float4 f1) {
  a[base + 0] = (_Float16)f0.x; a[base + 1] = (_Float16)f0.y;
  a[base + 2] = (_Float16)f0.z; a[base + 3] = (_Float16)f0.w;
  a[base + 4] = (_Float16)f1.x; a[base + 5] = (_Float16)f1.y;
  a[base + 6] = (_Float16)f1.z; a[base + 7] = (_Float16)f1.w;
}

// ---------------------------------------------------------------------------
// A-fragment (16x32 f16, CDNA5 ISA 7.12.2): lane<16 row M=lane holds
// K = ks*32+{0..7} (halves 0..7) and ks*32+{16..23} (halves 8..15);
// lanes 16..31 same rows with K offsets +8/+24.  rp = row pointer.
// KDIM % 8 == 0 always holds here (8, 64, 112), so 8-wide chunks are all-or-
// nothing: guards are branch-free clamped loads + cndmask zeroing.
// ---------------------------------------------------------------------------
template<int KDIM>
__device__ __forceinline__ v16h load_fragA(const float* __restrict__ rp, int ks, int lane) {
  const int kb = ks * 32 + ((lane >> 4) << 3);
  v16h a;
  if constexpr (KDIM % 32 == 0) {
    float4 f0 = *(const float4*)(rp + kb);
    float4 f1 = *(const float4*)(rp + kb + 4);
    float4 f2 = *(const float4*)(rp + kb + 16);
    float4 f3 = *(const float4*)(rp + kb + 20);
    cvt8(a, 0, f0, f1);
    cvt8(a, 8, f2, f3);
  } else {
    const bool v0 = (kb < KDIM);
    const bool v1 = (kb + 16 < KDIM);
    const int kb0 = v0 ? kb : 0;
    const int kb1 = v1 ? (kb + 16) : 0;
    float4 f0 = *(const float4*)(rp + kb0);
    float4 f1 = *(const float4*)(rp + kb0 + 4);
    float4 f2 = *(const float4*)(rp + kb1);
    float4 f3 = *(const float4*)(rp + kb1 + 4);
    const float4 z = make_float4(0.f, 0.f, 0.f, 0.f);
    if (!v0) { f0 = z; f1 = z; }
    if (!v1) { f2 = z; f3 = z; }
    cvt8(a, 0, f0, f1);
    cvt8(a, 8, f2, f3);
  }
  return a;
}

// ---------------------------------------------------------------------------
// Weight packer: W [KDIM x C] f32 row-major -> per-(kstep, ntile, lane)
// fragments of 16 contiguous f16 (B layout: lane<16 col N=n0+lane holds
// K=ks*32+{0..7,16..23}; lanes 16..31 hold K=ks*32+{8..15,24..31}).  Zero-pads K.
// ---------------------------------------------------------------------------
template<int KDIM, int C>
__global__ void pack_w_kernel(const float* __restrict__ W, _Float16* __restrict__ P) {
  constexpr int KS = (KDIM + 31) / 32;
  constexpr int NT = C / 16;
  const int t = blockIdx.x * blockDim.x + threadIdx.x;
  if (t >= KS * NT * 32) return;
  const int lane = t & 31;
  const int nt   = (t >> 5) % NT;
  const int ks   = t / (32 * NT);
  const int n    = nt * 16 + (lane & 15);
  const int kb   = ks * 32 + ((lane >> 4) << 3);
  _Float16* out = P + (size_t)t * 16;
#pragma unroll
  for (int j = 0; j < 16; ++j) {
    const int k = kb + (j & 7) + ((j >> 3) << 4);
    out[j] = (k < KDIM) ? (_Float16)W[(size_t)k * C + n] : (_Float16)0.0f;
  }
}

// ---------------------------------------------------------------------------
// Node projection: block = 128 threads = 4 waves; wave w computes output w
// of the same 16x16 tile:
//  0: q   = X@Wq  + bq          1: kn  = X@Wkn + bkn
//  2: vpl = X@Wv + bv + Y@Wlab + blab     3: out = X@Wsk + bsk (skip init)
// ---------------------------------------------------------------------------
template<int DIN, int C>
__global__ void __launch_bounds__(128)
proj_wmma_kernel(const float* __restrict__ X, const float* __restrict__ Y,
                 const _Float16* __restrict__ Pq, const _Float16* __restrict__ Pkn,
                 const _Float16* __restrict__ Pv, const _Float16* __restrict__ Plab,
                 const _Float16* __restrict__ Psk,
                 const float* __restrict__ bq, const float* __restrict__ bkn,
                 const float* __restrict__ bv, const float* __restrict__ blab,
                 const float* __restrict__ bsk,
                 float* __restrict__ qb, float* __restrict__ knb,
                 float* __restrict__ vplb, float* __restrict__ outb)
{
  constexpr int NT = C / 16;
  constexpr int KS = (DIN + 31) / 32;
  const int lane  = threadIdx.x & 31;
  const int which = threadIdx.x >> 5;
  const int row   = blockIdx.x * 16;     // N % 16 == 0
  const int nt    = blockIdx.y;
  const int n     = nt * 16 + (lane & 15);

  const _Float16* P1; const float* B1; float* D;
  if      (which == 0) { P1 = Pq;  B1 = bq;  D = qb;   }
  else if (which == 1) { P1 = Pkn; B1 = bkn; D = knb;  }
  else if (which == 2) { P1 = Pv;  B1 = bv;  D = vplb; }
  else                 { P1 = Psk; B1 = bsk; D = outb; }

  float binit = B1[n];
  if (which == 2) binit += blab[n];
  v8f acc;
#pragma unroll
  for (int r = 0; r < 8; ++r) acc[r] = binit;

  const float* rp = X + (size_t)(row + (lane & 15)) * DIN;
#pragma unroll
  for (int ks = 0; ks < KS; ++ks) {
    v16h a = load_fragA<DIN>(rp, ks, lane);
    v16h b = *(const v16h*)(P1 + ((size_t)(ks * NT + nt) * 32 + lane) * 16);
    acc = __builtin_amdgcn_wmma_f32_16x16x32_f16(false, a, false, b,
                                                 (short)0, acc, false, false);
  }
  if (which == 2) {  // + Y @ Wlab, K = 112 padded to 128
    const float* yp = Y + (size_t)(row + (lane & 15)) * 112;
#pragma unroll
    for (int ks = 0; ks < 4; ++ks) {
      v16h a = load_fragA<112>(yp, ks, lane);
      v16h b = *(const v16h*)(Plab + ((size_t)(ks * NT + nt) * 32 + lane) * 16);
      acc = __builtin_amdgcn_wmma_f32_16x16x32_f16(false, a, false, b,
                                                   (short)0, acc, false, false);
    }
  }

  const int mb = (lane >> 4) << 3;   // C/D: M = r + 8*(lane>=16), N = lane&15
#pragma unroll
  for (int r = 0; r < 8; ++r)
    D[(size_t)(row + mb + r) * C + n] = acc[r];
}

// ---------------------------------------------------------------------------
// Segment softmax helpers (ordered-uint float max encoding)
// ---------------------------------------------------------------------------
__device__ __forceinline__ unsigned encode_ord(float f) {
  unsigned b = __float_as_uint(f);
  return (b & 0x80000000u) ? ~b : (b | 0x80000000u);
}
__device__ __forceinline__ float decode_ord(unsigned u) {
  unsigned b = (u & 0x80000000u) ? (u & 0x7fffffffu) : ~u;
  return __uint_as_float(b);
}

__global__ void fill_stats_kernel(unsigned* __restrict__ smax, float* __restrict__ ssum, int n) {
  int i = blockIdx.x * blockDim.x + threadIdx.x;
  if (i < n) { smax[i] = 0u; ssum[i] = 0.0f; }  // 0 < encode(any finite)
}

// One wave per edge: score = q[dst] . (ke(edge) + kn[src]) / sqrt(C);
// ke recomputed on the fly from the 8-wide edge_attr (avoids an E*C buffer).
template<int C>
__global__ void __launch_bounds__(256)
score_kernel(const float* __restrict__ qb, const float* __restrict__ knb,
             const float* __restrict__ ea,
             const float* __restrict__ Wke, const float* __restrict__ bke,
             const int* __restrict__ src, const int* __restrict__ dst,
             float* __restrict__ scores, unsigned* __restrict__ smax,
             int E, float inv_sqrt_c)
{
  const int lane = threadIdx.x & 31;
  const int e = blockIdx.x * (blockDim.x >> 5) + (threadIdx.x >> 5);
  if (e >= E) return;
  const int s = src[e], d = dst[e];

  float eav[8];
#pragma unroll
  for (int t = 0; t < 8; ++t) eav[t] = ea[(size_t)e * 8 + t];

  float partial = 0.0f;
  constexpr int ITER = (C + 31) / 32;
#pragma unroll
  for (int i = 0; i < ITER; ++i) {
    const int c = lane + i * 32;
    if ((C % 32 == 0) || (c < C)) {
      float ke = bke[c];
#pragma unroll
      for (int t = 0; t < 8; ++t) ke = fmaf(eav[t], Wke[t * C + c], ke);
      partial = fmaf(qb[(size_t)d * C + c], ke + knb[(size_t)s * C + c], partial);
    }
  }
#pragma unroll
  for (int off = 16; off > 0; off >>= 1)
    partial += __shfl_xor(partial, off, 32);

  if (lane == 0) {
    const float sc = partial * inv_sqrt_c;
    scores[e] = sc;
    atomicMax(&smax[d], encode_ord(sc));
  }
}

// One thread per edge: e_val = exp(score - segmax[dst]); accumulate denom.
__global__ void exp_sum_kernel(float* __restrict__ scores,
                               const int* __restrict__ dst,
                               const unsigned* __restrict__ smax,
                               float* __restrict__ ssum, int E)
{
  int e = blockIdx.x * blockDim.x + threadIdx.x;
  if (e >= E) return;
  const int d = dst[e];
  const float m = decode_ord(smax[d]);
  const float ev = __expf(scores[e] - m);
  scores[e] = ev;
  __hip_atomic_fetch_add(&ssum[d], ev, __ATOMIC_RELAXED, __HIP_MEMORY_SCOPE_AGENT);
}

// One wave per edge: out[dst] += (v+lab)[src] * alpha
template<int C>
__global__ void __launch_bounds__(256)
aggregate_kernel(const float* __restrict__ vplb,
                 const float* __restrict__ scores,
                 const float* __restrict__ ssum,
                 const int* __restrict__ src, const int* __restrict__ dst,
                 float* __restrict__ out, int E)
{
  const int lane = threadIdx.x & 31;
  const int e = blockIdx.x * (blockDim.x >> 5) + (threadIdx.x >> 5);
  if (e >= E) return;
  const int s = src[e], d = dst[e];
  const float alpha = scores[e] / (ssum[d] + 1e-16f);
  constexpr int ITER = (C + 31) / 32;
#pragma unroll
  for (int i = 0; i < ITER; ++i) {
    const int c = lane + i * 32;
    if ((C % 32 == 0) || (c < C)) {
      __hip_atomic_fetch_add(&out[(size_t)d * C + c], vplb[(size_t)s * C + c] * alpha,
                             __ATOMIC_RELAXED, __HIP_MEMORY_SCOPE_AGENT);
    }
  }
}

__global__ void relu_kernel(float* __restrict__ h, size_t n) {
  size_t i = (size_t)blockIdx.x * blockDim.x + threadIdx.x;
  if (i < n) h[i] = fmaxf(h[i], 0.0f);
}

// ---------------------------------------------------------------------------
// Host side
// ---------------------------------------------------------------------------
template<int KDIM, int C>
static void launch_pack(const float* W, _Float16* P, hipStream_t s) {
  constexpr int total = ((KDIM + 31) / 32) * (C / 16) * 32;
  pack_w_kernel<KDIM, C><<<(total + 255) / 256, 256, 0, s>>>(W, P);
}

template<int DIN, int C>
static void run_layer(const float* Hin, const float* Y, const float* ea,
                      const int* src, const int* dst,
                      const float* Wq, const float* bq,
                      const float* Wke, const float* bke,
                      const float* Wkn, const float* bkn,
                      const float* Wv, const float* bv,
                      const float* Wlab, const float* blab,
                      const float* Wsk, const float* bsk,
                      _Float16* const* PW,   // 5 packed-weight buffers
                      float* qb, float* knb, float* vplb,
                      float* scores, float* ssum, unsigned* smax,
                      float* out, int do_relu, hipStream_t stream)
{
  const int N = NNODE, E = NEDGE;

  launch_pack<DIN, C>(Wq,  PW[0], stream);
  launch_pack<DIN, C>(Wkn, PW[1], stream);
  launch_pack<DIN, C>(Wv,  PW[2], stream);
  launch_pack<112, C>(Wlab, PW[3], stream);
  launch_pack<DIN, C>(Wsk, PW[4], stream);

  fill_stats_kernel<<<(N + 255) / 256, 256, 0, stream>>>(smax, ssum, N);

  dim3 pgrid(N / 16, C / 16);
  proj_wmma_kernel<DIN, C><<<pgrid, 128, 0, stream>>>(
      Hin, Y, PW[0], PW[1], PW[2], PW[3], PW[4],
      bq, bkn, bv, blab, bsk, qb, knb, vplb, out);

  const float inv_sqrt_c = 1.0f / sqrtf((float)C);
  const int edge_wave_blocks = (E + 7) / 8;  // 8 waves of 32 per block
  score_kernel<C><<<edge_wave_blocks, 256, 0, stream>>>(
      qb, knb, ea, Wke, bke, src, dst, scores, smax, E, inv_sqrt_c);

  exp_sum_kernel<<<(E + 255) / 256, 256, 0, stream>>>(scores, dst, smax, ssum, E);

  aggregate_kernel<C><<<edge_wave_blocks, 256, 0, stream>>>(
      vplb, scores, ssum, src, dst, out, E);

  if (do_relu) {
    size_t tot = (size_t)N * C;
    relu_kernel<<<(unsigned)((tot + 255) / 256), 256, 0, stream>>>(out, tot);
  }
}

extern "C" void kernel_launch(void* const* d_in, const int* in_sizes, int n_in,
                              void* d_out, int out_size, void* d_ws, size_t ws_size,
                              hipStream_t stream) {
  const int N = NNODE, E = NEDGE;
  const float* x  = (const float*)d_in[0];
  const int*   ei = (const int*)d_in[1];
  const float* ea = (const float*)d_in[2];
  const float* y  = (const float*)d_in[3];
  const int* src = ei;
  const int* dst = ei + E;

  // params flattened per layer in dict insertion order:
  // W_q,b_q, W_ke,b_ke, W_kn,b_kn, W_kl,b_kl, W_v,b_v, W_lab,b_lab, W_skip,b_skip
  auto P = [&](int layer, int idx) -> const float* {
    return (const float*)d_in[4 + layer * 14 + idx];
  };

  float* ws     = (float*)d_ws;
  float* qb     = ws;                              // N*112 (max C)
  float* knb    = qb   + (size_t)N * 112;          // N*112
  float* vplb   = knb  + (size_t)N * 112;          // N*112
  float* h1     = vplb + (size_t)N * 112;          // N*64
  float* h2     = h1   + (size_t)N * 64;           // N*64
  float* scores = h2   + (size_t)N * 64;           // E
  float* ssum   = scores + (size_t)E;              // N
  unsigned* smax = (unsigned*)(ssum + N);          // N
  // packed f16 weights: 5 buffers, each up to (128/32)*(112/16)*32*16 = 14336
  // halves; round to 16384 halves (32 KB) apiece.
  _Float16* pbase = (_Float16*)(smax + N);
  _Float16* PW[5];
  for (int i = 0; i < 5; ++i) PW[i] = pbase + (size_t)i * 16384;

  float* out = (float*)d_out;                      // N*112

  // Layer 0: 8 -> 64, ReLU
  run_layer<8, 64>(x, y, ea, src, dst,
                   P(0,0), P(0,1), P(0,2), P(0,3), P(0,4), P(0,5),
                   P(0,8), P(0,9), P(0,10), P(0,11), P(0,12), P(0,13),
                   PW, qb, knb, vplb, scores, ssum, smax, h1, 1, stream);

  // Layer 1: 64 -> 64, ReLU
  run_layer<64, 64>(h1, y, ea, src, dst,
                    P(1,0), P(1,1), P(1,2), P(1,3), P(1,4), P(1,5),
                    P(1,8), P(1,9), P(1,10), P(1,11), P(1,12), P(1,13),
                    PW, qb, knb, vplb, scores, ssum, smax, h2, 1, stream);

  // Layer 2: 64 -> 112, no ReLU, writes d_out
  run_layer<64, 112>(h2, y, ea, src, dst,
                     P(2,0), P(2,1), P(2,2), P(2,3), P(2,4), P(2,5),
                     P(2,8), P(2,9), P(2,10), P(2,11), P(2,12), P(2,13),
                     PW, qb, knb, vplb, scores, ssum, smax, out, 0, stream);
}
